// MyGRUCell_472446402807
// MI455X (gfx1250) — matrix operand order repaired and checked by reading, not compiled, not run on hardware
//
#include <hip/hip_runtime.h>
#include <hip/hip_bf16.h>

// GRU cell for MI455X (gfx1250, wave32).
// Compute-bound bf16 WMMA (v_wmma_f32_16x16x32_bf16, fp32 accumulate).
// L2-traffic-bound at wave-private loads -> stage 128x128 block tiles in LDS
// via async-to-LDS copies (ASYNCcnt) with double buffering; consume with
// conflict-free ds_load_b128 fragment reads.

typedef __attribute__((ext_vector_type(16))) __bf16 v16bf;
typedef __attribute__((ext_vector_type(8)))  __bf16 v8bf;
typedef __attribute__((ext_vector_type(8)))  float  v8f;

#define B_ROWS 8192
#define I_DIM  1024
#define H_DIM  1024
#define K_DIM  2048        // I + H
#define KCHUNK 64
#define NCHUNKS (K_DIM / KCHUNK)   // 32 (16 per A half)
#define LDS_STRIDE 72      // 64 + 8 bf16 pad -> 144B row stride, bank-conflict-free

static __device__ __forceinline__ v16bf cat8(v8bf lo, v8bf hi) {
  return __builtin_shufflevector(lo, hi, 0,1,2,3,4,5,6,7,8,9,10,11,12,13,14,15);
}

// Fast activations: branch-free, hardware v_rcp_f32 / v_exp_f32.
static __device__ __forceinline__ float fast_sigmoid(float x) {
  return __builtin_amdgcn_rcpf(1.0f + __expf(-x));
}
static __device__ __forceinline__ float fast_tanh(float x) {
  return 1.0f - 2.0f * __builtin_amdgcn_rcpf(1.0f + __expf(2.0f * x));
}

// 16B global -> LDS async copy (tracked by ASYNCcnt).
// LDS byte address = low 32 bits of the generic pointer (LDS aperture rule).
static __device__ __forceinline__
void async_copy16(const __bf16* gptr, const __bf16* lptr) {
  uint32_t laddr = (uint32_t)(uintptr_t)lptr;
  asm volatile("global_load_async_to_lds_b128 %0, %1, off"
               :: "v"(laddr), "v"(gptr)
               : "memory");
}

static __device__ __forceinline__ void wait_async_le8() {
  asm volatile("s_wait_asynccnt 0x8" ::: "memory");
}
static __device__ __forceinline__ void wait_async_0() {
  asm volatile("s_wait_asynccnt 0x0" ::: "memory");
}

// ---------------------------------------------------------------------------
// f32 -> bf16 conversion, 8 elements / thread.
// ---------------------------------------------------------------------------
__global__ __launch_bounds__(256)
void gru_cvt_f32_bf16(const float* __restrict__ in, __bf16* __restrict__ out, int n8) {
  int i = blockIdx.x * blockDim.x + threadIdx.x;
  if (i >= n8) return;
  const float4* p = (const float4*)in + 2 * i;
  float4 f0 = p[0];
  float4 f1 = p[1];
  v8bf o;
  o[0] = (__bf16)f0.x; o[1] = (__bf16)f0.y; o[2] = (__bf16)f0.z; o[3] = (__bf16)f0.w;
  o[4] = (__bf16)f1.x; o[5] = (__bf16)f1.y; o[6] = (__bf16)f1.z; o[7] = (__bf16)f1.w;
  *((v8bf*)out + i) = o;
}

// ---------------------------------------------------------------------------
// WMMA fragment set (CDNA5 ISA 7.12.2 wave32 layouts), read from LDS:
//   A (16x32): lane l (l16=l&15, half=l>>4) holds row m+l16,
//              K chunks [s*32+8*half,+8) and [s*32+16+8*half,+8).
//   B (32x16): lane l holds column n+l16,
//              K chunks [s*32+8*half,+8) and [s*32+16+8*half,+8).
// ---------------------------------------------------------------------------
struct FragSet {
  v16bf a[2];
  v16bf b[4];
};

static __device__ __forceinline__
void do_wmma8(v8f acc[2][4], const FragSet& f) {
#pragma unroll
  for (int mt = 0; mt < 2; ++mt)
#pragma unroll
    for (int nt = 0; nt < 4; ++nt)
      acc[mt][nt] = __builtin_amdgcn_wmma_f32_16x16x32_bf16(
          false, f.a[mt], false, f.b[nt], (short)0, acc[mt][nt], false, false);
}

// ---------------------------------------------------------------------------
// LDS-staged block GEMM mainloop.
// Block tile: 128(M) x 128(N), K-chunk 64, 8 waves as 4(M) x 2(N),
// wave tile 32x64 (acc[2][4] of v8f).
// Double-buffered async copies: issue chunk kc+1, wait ASYNCcnt<=8 (the 8
// just-issued stay in flight; async loads complete in order), barrier,
// consume chunk kc from LDS, barrier (protects buffer reuse at kc+2).
// ---------------------------------------------------------------------------
static __device__ __forceinline__
void wmma_mainloop_lds(const __bf16* __restrict__ A0,   // K in [0,1024)
                       const __bf16* __restrict__ A1,   // K in [1024,2048)
                       const __bf16* __restrict__ W,
                       int mblk, int nblk,
                       __bf16 (*As)[128][LDS_STRIDE],
                       __bf16 (*Ws)[128][LDS_STRIDE],
                       v8f acc[2][4]) {
  const int t    = threadIdx.x;
  const int lane = t & 31;
  const int wave = t >> 5;
  const int wm   = wave >> 1;          // 0..3 -> M
  const int wn   = wave & 1;           // 0..1 -> N
  const int half = lane >> 4;
  const int l16  = lane & 15;

  // Copy assignment: thread t moves rows {cpr, cpr+32, cpr+64, cpr+96},
  // 16B column slice cpc, for both A and W tiles (8 async ops / chunk).
  const int cpr = t >> 3;              // 0..31
  const int cpc = (t & 7) * 8;         // 0,8,...,56 (bf16 elements)

  // Issue one 64-deep K chunk into buffer `buf`.
  auto issue = [&](int kc, int buf) {
    const __bf16* ab;
    int ac;
    if (kc < NCHUNKS / 2) { ab = A0; ac = kc * KCHUNK; }
    else                  { ab = A1; ac = (kc - NCHUNKS / 2) * KCHUNK; }
    const int kw = kc * KCHUNK;
#pragma unroll
    for (int i = 0; i < 4; ++i) {
      const int r = cpr + 32 * i;
      async_copy16(ab + (mblk + r) * I_DIM + ac + cpc, &As[buf][r][cpc]);
    }
#pragma unroll
    for (int i = 0; i < 4; ++i) {
      const int r = cpr + 32 * i;
      async_copy16(W + (nblk + r) * K_DIM + kw + cpc, &Ws[buf][r][cpc]);
    }
  };

  // Consume one 64-deep K chunk from buffer `buf` (two 32-deep WMMA steps).
  auto consume = [&](int buf) {
#pragma unroll
    for (int s = 0; s < 2; ++s) {
      FragSet f;
#pragma unroll
      for (int mt = 0; mt < 2; ++mt) {
        const __bf16* p = &As[buf][wm * 32 + mt * 16 + l16][s * 32 + 8 * half];
        f.a[mt] = cat8(*(const v8bf*)p, *(const v8bf*)(p + 16));
      }
#pragma unroll
      for (int nt = 0; nt < 4; ++nt) {
        const __bf16* p = &Ws[buf][wn * 64 + nt * 16 + l16][s * 32 + 8 * half];
        f.b[nt] = cat8(*(const v8bf*)p, *(const v8bf*)(p + 16));
      }
      do_wmma8(acc, f);
    }
  };

  issue(0, 0);
  for (int kc = 0; kc < NCHUNKS - 1; ++kc) {
    issue(kc + 1, (kc + 1) & 1);
    wait_async_le8();          // chunk kc complete; kc+1 still in flight
    __syncthreads();
    consume(kc & 1);
    __syncthreads();           // all waves done reading before kc+2 overwrites
  }
  wait_async_0();              // last chunk (no overlap left)
  __syncthreads();
  consume((NCHUNKS - 1) & 1);
}

// ---------------------------------------------------------------------------
// Phase 2: r and z gates (compile-time GATE -> straight-line epilogues).
//   GATE 0: rh = bf16( sigmoid(acc + br) * h_prev )   (feeds phase-3 A matrix)
//   GATE 1: z  = sigmoid(acc + bz)                    (f32, for final combine)
// ---------------------------------------------------------------------------
template <int GATE>
__global__ __launch_bounds__(256)
void gru_gemm_rz(const __bf16* __restrict__ Xb, const __bf16* __restrict__ Hb,
                 const __bf16* __restrict__ W, const float* __restrict__ bias,
                 const float* __restrict__ h_prev,
                 __bf16* __restrict__ RHb, float* __restrict__ Zf) {
  __shared__ __bf16 As[2][128][LDS_STRIDE];
  __shared__ __bf16 Ws[2][128][LDS_STRIDE];

  const int lane = threadIdx.x & 31;
  const int wave = threadIdx.x >> 5;
  const int wm = wave >> 1;
  const int wn = wave & 1;
  const int mblk = blockIdx.x * 128;
  const int nblk = blockIdx.y * 128;

  v8f zero = {};
  v8f acc[2][4];
#pragma unroll
  for (int mt = 0; mt < 2; ++mt)
#pragma unroll
    for (int nt = 0; nt < 4; ++nt) acc[mt][nt] = zero;

  wmma_mainloop_lds(Xb, Hb, W, mblk, nblk, As, Ws, acc);

  const int half = lane >> 4;
  const int l16  = lane & 15;
  const int m0 = mblk + wm * 32;
  const int n0 = nblk + wn * 64;
#pragma unroll
  for (int mt = 0; mt < 2; ++mt) {
#pragma unroll
    for (int nt = 0; nt < 4; ++nt) {
      const int col = n0 + nt * 16 + l16;
      const float bv = bias[col];
#pragma unroll
      for (int i = 0; i < 8; ++i) {
        const int row = m0 + mt * 16 + half * 8 + i;   // C/D layout: VGPR i -> M=i(+8)
        const int idx = row * H_DIM + col;
        const float s = fast_sigmoid(acc[mt][nt][i] + bv);
        if (GATE == 0) RHb[idx] = (__bf16)(s * h_prev[idx]);
        else           Zf[idx]  = s;
      }
    }
  }
}

// ---------------------------------------------------------------------------
// Phase 3: candidate g + final combine.
//   g = tanh([x | r*h] @ Wg^T + bg);  h_new = (1-z)*g + z*h_prev
// ---------------------------------------------------------------------------
__global__ __launch_bounds__(256)
void gru_gemm_g(const __bf16* __restrict__ Xb, const __bf16* __restrict__ RHb,
                const __bf16* __restrict__ Wgb, const float* __restrict__ bg,
                const float* __restrict__ Zf, const float* __restrict__ h_prev,
                float* __restrict__ out) {
  __shared__ __bf16 As[2][128][LDS_STRIDE];
  __shared__ __bf16 Ws[2][128][LDS_STRIDE];

  const int lane = threadIdx.x & 31;
  const int wave = threadIdx.x >> 5;
  const int wm = wave >> 1;
  const int wn = wave & 1;
  const int mblk = blockIdx.x * 128;
  const int nblk = blockIdx.y * 128;

  v8f zero = {};
  v8f acc[2][4];
#pragma unroll
  for (int mt = 0; mt < 2; ++mt)
#pragma unroll
    for (int nt = 0; nt < 4; ++nt) acc[mt][nt] = zero;

  wmma_mainloop_lds(Xb, RHb, Wgb, mblk, nblk, As, Ws, acc);

  const int half = lane >> 4;
  const int l16  = lane & 15;
  const int m0 = mblk + wm * 32;
  const int n0 = nblk + wn * 64;
#pragma unroll
  for (int mt = 0; mt < 2; ++mt) {
#pragma unroll
    for (int nt = 0; nt < 4; ++nt) {
      const int col = n0 + nt * 16 + l16;
      const float bv = bg[col];
#pragma unroll
      for (int i = 0; i < 8; ++i) {
        const int row = m0 + mt * 16 + half * 8 + i;
        const int idx = row * H_DIM + col;
        const float g = fast_tanh(acc[mt][nt][i] + bv);
        const float z = Zf[idx];
        out[idx] = (1.0f - z) * g + z * h_prev[idx];
      }
    }
  }
}

// ---------------------------------------------------------------------------
// Host side.
// Workspace layout (bytes):          size
//   Xb  (bf16, 8192x1024)     @ 0     16 MB
//   Hb  (bf16, 8192x1024)     @ 16MB  16 MB
//   Wrb (bf16, 1024x2048)     @ 32MB   4 MB
//   Wzb (bf16, 1024x2048)     @ 36MB   4 MB
//   Wgb (bf16, 1024x2048)     @ 40MB   4 MB
//   RHb (bf16, 8192x1024)     @ 44MB  16 MB
//   Zf  (f32,  8192x1024)     @ 60MB  32 MB
// total: 92 MB
// ---------------------------------------------------------------------------
extern "C" void kernel_launch(void* const* d_in, const int* in_sizes, int n_in,
                              void* d_out, int out_size, void* d_ws, size_t ws_size,
                              hipStream_t stream) {
  const float* x  = (const float*)d_in[0];
  const float* h  = (const float*)d_in[1];
  const float* Wr = (const float*)d_in[2];
  const float* br = (const float*)d_in[3];
  const float* Wz = (const float*)d_in[4];
  const float* bz = (const float*)d_in[5];
  const float* Wg = (const float*)d_in[6];
  const float* bg = (const float*)d_in[7];
  float* out = (float*)d_out;

  char* ws = (char*)d_ws;
  const size_t MB = 1u << 20;
  __bf16* Xb  = (__bf16*)(ws + 0 * MB);
  __bf16* Hb  = (__bf16*)(ws + 16 * MB);
  __bf16* Wrb = (__bf16*)(ws + 32 * MB);
  __bf16* Wzb = (__bf16*)(ws + 36 * MB);
  __bf16* Wgb = (__bf16*)(ws + 40 * MB);
  __bf16* RHb = (__bf16*)(ws + 44 * MB);
  float*  Zf  = (float*)(ws + 60 * MB);

  // ---- phase 1: f32 -> bf16 conversions -----------------------------------
  const int nXH8 = (B_ROWS * I_DIM) / 8;
  const int nW8  = (H_DIM * K_DIM) / 8;
  gru_cvt_f32_bf16<<<nXH8 / 256, 256, 0, stream>>>(x,  Xb,  nXH8);
  gru_cvt_f32_bf16<<<nXH8 / 256, 256, 0, stream>>>(h,  Hb,  nXH8);
  gru_cvt_f32_bf16<<<nW8  / 256, 256, 0, stream>>>(Wr, Wrb, nW8);
  gru_cvt_f32_bf16<<<nW8  / 256, 256, 0, stream>>>(Wz, Wzb, nW8);
  gru_cvt_f32_bf16<<<nW8  / 256, 256, 0, stream>>>(Wg, Wgb, nW8);

  // ---- phase 2: r & z gates -----------------------------------------------
  dim3 grid_rz(B_ROWS / 128, H_DIM / 128, 1);
  gru_gemm_rz<0><<<grid_rz, 256, 0, stream>>>(Xb, Hb, Wrb, br, h, RHb, Zf);
  gru_gemm_rz<1><<<grid_rz, 256, 0, stream>>>(Xb, Hb, Wzb, bz, h, RHb, Zf);

  // ---- phase 3: candidate + combine ---------------------------------------
  dim3 grid_g(B_ROWS / 128, H_DIM / 128, 1);
  gru_gemm_g<<<grid_g, 256, 0, stream>>>(Xb, RHb, Wgb, bg, Zf, h, out);
}
